// PointTransformerLayer_54631984005333
// MI455X (gfx1250) — matrix operand order repaired and checked by reading, not compiled, not run on hardware
//
#include <hip/hip_runtime.h>

// ---------------------------------------------------------------------------
// Point Transformer layer for MI455X (gfx1250, wave32, WMMA).
// B=1, N=1024, C_IN=128, C_OUT=256, H=8, D=32, FFN=1024.
// Key algebraic optimization: pe.reshape(..,H,D).mean(-1) commutes with the
// second pos-MLP linear => pre-reduce Wp2 to (256 x 8), cutting 137 GFLOP
// to ~4 GFLOP. All GEMMs run on v_wmma_f32_16x16x32_bf16 with f32 accum.
// bf16 conversion uses the native v_cvt path ((__bf16) cast) instead of
// manual integer rounding; long-K GEMMs prefetch one K-tile ahead.
// ---------------------------------------------------------------------------

typedef __attribute__((ext_vector_type(16))) __bf16 v16bf;
typedef __attribute__((ext_vector_type(8)))  float  v8f;

union FragU {
  uint4 q[2];
  unsigned short s[16];
  __bf16 h[16];
  v16bf bv;
};

__device__ __forceinline__ unsigned short f2b(float f) {
  return __builtin_bit_cast(unsigned short, (__bf16)f);   // native v_cvt, RNE
}

// A-matrix fragment (M x K, row held within a lane).
// lane-half h2: K offsets {8h2..8h2+7} U {16+8h2..16+8h2+7} from `base` (contiguous K).
__device__ __forceinline__ v16bf load_frag_a(const unsigned short* base, int h2) {
  FragU f;
  f.q[0] = *(const uint4*)(base + 8 * h2);
  f.q[1] = *(const uint4*)(base + 16 + 8 * h2);
  return f.bv;
}

// B-matrix fragment (K x N, row striped across lanes): lane n holds column n,
// lane-half h2: K offsets {8h2..8h2+7} U {16+8h2..16+8h2+7}. `base` = column-n data, contiguous K.
__device__ __forceinline__ v16bf load_frag_b(const unsigned short* base, int h2) {
  FragU f;
  f.q[0] = *(const uint4*)(base + 8 * h2);
  f.q[1] = *(const uint4*)(base + 16 + 8 * h2);
  return f.bv;
}

__device__ __forceinline__ v8f wmma_bf16(v16bf a, v16bf b, v8f c) {
  // 8-arg form: (neg_a, A, neg_b, B, c_mod, C, reuse_a, reuse_b)
  return __builtin_amdgcn_wmma_f32_16x16x32_bf16(false, a, false, b, (short)0, c,
                                                 false, false);
}

struct InPtrs { const float* p[24]; };

// ---------------------------------------------------------------------------
// Kernel 0: weight conversion / transposition / Wp2 head-mean reduction.
// ---------------------------------------------------------------------------
__global__ void k_prep(InPtrs in,
                       unsigned short* wqkvr_t, float* bqkvr,
                       unsigned short* wo_t, unsigned short* wf1_t,
                       unsigned short* wf2_t, unsigned short* wp2h_t,
                       float* bp2h, unsigned short* xb) {
  const float* x   = in.p[0];
  const float* Wq  = in.p[2];  const float* bq  = in.p[3];
  const float* Wk  = in.p[4];  const float* bk  = in.p[5];
  const float* Wv  = in.p[6];  const float* bv  = in.p[7];
  const float* Wp2 = in.p[10]; const float* bp2 = in.p[11];
  const float* Wo  = in.p[12];
  const float* Wr  = in.p[14]; const float* br  = in.p[15];
  const float* Wf1 = in.p[20];
  const float* Wf2 = in.p[22];

  const int T0 = 1024 * 128;       // wqkvr_t  (row o=0..1023, col c=0..127)
  const int T1 = T0 + 1024;        // bqkvr
  const int T2 = T1 + 256 * 256;   // wo_t
  const int T3 = T2 + 1024 * 256;  // wf1_t
  const int T4 = T3 + 256 * 1024;  // wf2_t
  const int T5 = T4 + 16 * 256;    // wp2h_t (rows 8..15 zero-padded)
  const int T6 = T5 + 16;          // bp2h
  const int T7 = T6 + 1024 * 128;  // xb

  for (int idx = blockIdx.x * blockDim.x + threadIdx.x; idx < T7;
       idx += gridDim.x * blockDim.x) {
    if (idx < T0) {
      int o = idx >> 7, c = idx & 127;
      const float* W = (o < 256) ? Wq : (o < 512) ? Wk : (o < 768) ? Wv : Wr;
      wqkvr_t[idx] = f2b(W[c * 256 + (o & 255)]);
    } else if (idx < T1) {
      int o = idx - T0;
      const float* B = (o < 256) ? bq : (o < 512) ? bk : (o < 768) ? bv : br;
      bqkvr[o] = B[o & 255];
    } else if (idx < T2) {
      int i2 = idx - T1; int o = i2 >> 8, c = i2 & 255;
      wo_t[i2] = f2b(Wo[c * 256 + o]);
    } else if (idx < T3) {
      int i2 = idx - T2; int o = i2 >> 8, c = i2 & 255;
      wf1_t[i2] = f2b(Wf1[c * 1024 + o]);
    } else if (idx < T4) {
      int i2 = idx - T3; int o = i2 >> 10, c = i2 & 1023;
      wf2_t[i2] = f2b(Wf2[c * 256 + o]);
    } else if (idx < T5) {
      int i2 = idx - T4; int h = i2 >> 8, c = i2 & 255;
      float s = 0.f;
      if (h < 8) {
        for (int d = 0; d < 32; ++d) s += Wp2[c * 256 + h * 32 + d];
        s *= (1.f / 32.f);
      }
      wp2h_t[i2] = f2b(s);
    } else if (idx < T6) {
      int h = idx - T5;
      float s = 0.f;
      if (h < 8) {
        for (int d = 0; d < 32; ++d) s += bp2[h * 32 + d];
        s *= (1.f / 32.f);
      }
      bp2h[h] = s;
    } else {
      int i2 = idx - T6;
      xb[i2] = f2b(x[i2]);
    }
  }
}

// ---------------------------------------------------------------------------
// Kernel 1: positional-encoding bias, written directly into S[h][i][j].
// Each wave: 16 (i,j) pairs x 8 heads via one 16x16x256 bf16 GEMM
// (A = relu(rel@Wp1+bp1) built on-the-fly in VALU, B = reduced Wp2h^T).
// ---------------------------------------------------------------------------
__global__ void k_pe(const float* pos, const float* Wp1, const float* bp1,
                     const unsigned short* wp2h_t, const float* bp2h, float* S) {
  int w    = blockIdx.x * (blockDim.x >> 5) + (threadIdx.x >> 5);
  int lane = threadIdx.x & 31;
  int h2 = lane >> 4, m = lane & 15;
  int pbase = w * 16;
  int i  = pbase >> 10;
  int j0 = pbase & 1023;
  int j  = j0 + m;                 // A-row m <-> pair (i, j0+m)
  float dx = pos[i * 3 + 0] - pos[j * 3 + 0];
  float dy = pos[i * 3 + 1] - pos[j * 3 + 1];
  float dz = pos[i * 3 + 2] - pos[j * 3 + 2];

  v8f acc = {};
  for (int kt = 0; kt < 8; ++kt) {
    int k0 = kt * 32;
    FragU a;
#pragma unroll
    for (int t = 0; t < 16; ++t) {
      int c = k0 + 8 * h2 + (t & 7) + ((t >> 3) << 4);
      float hv = bp1[c] + dx * Wp1[c] + dy * Wp1[256 + c] + dz * Wp1[512 + c];
      a.h[t] = (__bf16)(hv > 0.f ? hv : 0.f);   // native cvt
    }
    v16bf b = load_frag_b(wp2h_t + m * 256 + k0, h2);  // B col n=m (head)
    acc = wmma_bf16(a.bv, b, acc);
  }
  if (m < 8) {                      // valid head columns 0..7
    float bb = bp2h[m];
#pragma unroll
    for (int e = 0; e < 8; ++e) {
      int jj = j0 + e + 8 * h2;     // pair row e+8h2
      S[((size_t)(m * 1024 + i) << 10) + jj] = acc[e] + bb;
    }
  }
}

// ---------------------------------------------------------------------------
// Kernel 2: fused x @ [Wq|Wk|Wv|Wr] + bias.  q,k bf16 row-major; v bf16
// transposed (for attn@v B-fragments); identity fp32.
// ---------------------------------------------------------------------------
__global__ void k_qkvr(const unsigned short* xb, const unsigned short* wqkvr_t,
                       const float* bqkvr, unsigned short* qb, unsigned short* kb,
                       unsigned short* vt, float* ident) {
  int w    = blockIdx.x * (blockDim.x >> 5) + (threadIdx.x >> 5);
  int lane = threadIdx.x & 31;
  int h2 = lane >> 4, nm = lane & 15;
  int mt = w >> 6, nt = w & 63;
  int rowbase = mt * 16, colbase = nt * 16;
  v8f acc = {};
  for (int kt = 0; kt < 4; ++kt) {
    int k0 = kt * 32;
    v16bf a = load_frag_a(xb + (rowbase + nm) * 128 + k0, h2);
    v16bf b = load_frag_b(wqkvr_t + (colbase + nm) * 128 + k0, h2);
    acc = wmma_bf16(a, b, acc);
  }
  int o = colbase + nm;
  float bias = bqkvr[o];
#pragma unroll
  for (int e = 0; e < 8; ++e) {
    int r = rowbase + e + 8 * h2;
    float val = acc[e] + bias;
    if (o < 256)      qb[r * 256 + o] = f2b(val);
    else if (o < 512) kb[r * 256 + (o - 256)] = f2b(val);
    else if (o < 768) vt[(o - 512) * 1024 + r] = f2b(val);   // transposed
    else              ident[r * 256 + (o - 768)] = val;
  }
}

// ---------------------------------------------------------------------------
// Kernel 3: S += (q . k^T) / sqrt(D) per head.  K=32 -> exactly one WMMA/tile.
// ---------------------------------------------------------------------------
__global__ void k_qk(const unsigned short* qb, const unsigned short* kb, float* S) {
  int w    = blockIdx.x * (blockDim.x >> 5) + (threadIdx.x >> 5);
  int lane = threadIdx.x & 31;
  int h2 = lane >> 4, nm = lane & 15;
  int h = w >> 12, rem = w & 4095;
  int mt = rem >> 6, nt = rem & 63;
  int rowbase = mt * 16, colbase = nt * 16;
  v16bf a = load_frag_a(qb + (rowbase + nm) * 256 + h * 32, h2);
  v16bf b = load_frag_b(kb + (colbase + nm) * 256 + h * 32, h2);
  v8f acc = {};
  acc = wmma_bf16(a, b, acc);
  const float scale = 0.17677669529663687f;  // 1/sqrt(32)
#pragma unroll
  for (int e = 0; e < 8; ++e) {
    size_t idx = ((size_t)(h * 1024 + rowbase + e + 8 * h2) << 10) + colbase + nm;
    S[idx] += acc[e] * scale;
  }
}

// ---------------------------------------------------------------------------
// Kernel 4: row softmax over j (length 1024) per (head, query); bf16 output.
// ---------------------------------------------------------------------------
__global__ void k_softmax(const float* S, unsigned short* attnb) {
  __shared__ float sh[256];
  int t = threadIdx.x;
  int h = blockIdx.x >> 10, i = blockIdx.x & 1023;
  size_t base = ((size_t)(h * 1024 + i)) << 10;
  float v[4];
  float mx = -3.4e38f;
#pragma unroll
  for (int k = 0; k < 4; ++k) { v[k] = S[base + t + (k << 8)]; mx = fmaxf(mx, v[k]); }
  sh[t] = mx; __syncthreads();
  for (int s2 = 128; s2 > 0; s2 >>= 1) {
    if (t < s2) sh[t] = fmaxf(sh[t], sh[t + s2]);
    __syncthreads();
  }
  mx = sh[0]; __syncthreads();
  float sum = 0.f;
#pragma unroll
  for (int k = 0; k < 4; ++k) { v[k] = __expf(v[k] - mx); sum += v[k]; }
  sh[t] = sum; __syncthreads();
  for (int s2 = 128; s2 > 0; s2 >>= 1) {
    if (t < s2) sh[t] += sh[t + s2];
    __syncthreads();
  }
  float inv = 1.f / sh[0];
#pragma unroll
  for (int k = 0; k < 4; ++k) attnb[base + t + (k << 8)] = f2b(v[k] * inv);
}

// ---------------------------------------------------------------------------
// Kernel 5: out = attn @ v per head (K=1024 -> 32 WMMAs per 16x16 tile).
// Prefetch one K-tile ahead (global_prefetch_b8) to hide L2 latency.
// ---------------------------------------------------------------------------
__global__ void k_av(const unsigned short* attnb, const unsigned short* vt,
                     unsigned short* ob) {
  int w    = blockIdx.x * (blockDim.x >> 5) + (threadIdx.x >> 5);
  int lane = threadIdx.x & 31;
  int h2 = lane >> 4, nm = lane & 15;
  int h = w >> 7, rem = w & 127;
  int mt = rem >> 1, ntile = rem & 1;
  int rowbase = mt * 16;
  const unsigned short* arow = attnb + (((size_t)(h * 1024 + rowbase + nm)) << 10);
  const unsigned short* brow = vt + (h * 32 + ntile * 16 + nm) * 1024;
  v8f acc = {};
  for (int kt = 0; kt < 32; ++kt) {
    int k0 = kt * 32;
    if (kt + 1 < 32) {
      __builtin_prefetch(arow + k0 + 32, 0, 0);
      __builtin_prefetch(brow + k0 + 32, 0, 0);
    }
    v16bf a = load_frag_a(arow + k0, h2);
    v16bf b = load_frag_b(brow + k0, h2);
    acc = wmma_bf16(a, b, acc);
  }
  int o = h * 32 + ntile * 16 + nm;
#pragma unroll
  for (int e = 0; e < 8; ++e) {
    int r = rowbase + e + 8 * h2;
    ob[r * 256 + o] = f2b(acc[e]);
  }
}

// ---------------------------------------------------------------------------
// Kernel 6: out @ Wo + bo + identity -> x1raw (pre-LN1) fp32.
// ---------------------------------------------------------------------------
__global__ void k_oproj(const unsigned short* ob, const unsigned short* wo_t,
                        const float* bo, const float* ident, float* x1raw) {
  int w    = blockIdx.x * (blockDim.x >> 5) + (threadIdx.x >> 5);
  int lane = threadIdx.x & 31;
  int h2 = lane >> 4, nm = lane & 15;
  int mt = w >> 4, nt = w & 15;
  int rowbase = mt * 16, colbase = nt * 16;
  v8f acc = {};
  for (int kt = 0; kt < 8; ++kt) {
    int k0 = kt * 32;
    v16bf a = load_frag_a(ob + (rowbase + nm) * 256 + k0, h2);
    v16bf b = load_frag_b(wo_t + (colbase + nm) * 256 + k0, h2);
    acc = wmma_bf16(a, b, acc);
  }
  int o = colbase + nm;
  float bias = bo[o];
#pragma unroll
  for (int e = 0; e < 8; ++e) {
    int r = rowbase + e + 8 * h2;
    x1raw[r * 256 + o] = acc[e] + bias + ident[r * 256 + o];
  }
}

// ---------------------------------------------------------------------------
// Kernel 7 (used twice): row LayerNorm over 256 channels; optional bf16 copy.
// ---------------------------------------------------------------------------
__global__ void k_ln(const float* xin, const float* g, const float* b,
                     float* yout, unsigned short* ybf) {
  __shared__ float sh[256];
  int t = threadIdx.x, r = blockIdx.x;
  float v = xin[r * 256 + t];
  sh[t] = v; __syncthreads();
  for (int s2 = 128; s2 > 0; s2 >>= 1) {
    if (t < s2) sh[t] += sh[t + s2];
    __syncthreads();
  }
  float mean = sh[0] * (1.f / 256.f); __syncthreads();
  float d = v - mean;
  sh[t] = d * d; __syncthreads();
  for (int s2 = 128; s2 > 0; s2 >>= 1) {
    if (t < s2) sh[t] += sh[t + s2];
    __syncthreads();
  }
  float var = sh[0] * (1.f / 256.f); __syncthreads();
  float y = d * rsqrtf(var + 1e-5f) * g[t] + b[t];
  yout[r * 256 + t] = y;
  if (ybf) ybf[r * 256 + t] = f2b(y);
}

// ---------------------------------------------------------------------------
// Kernel 8: FFN first linear + ReLU, bf16 output.
// ---------------------------------------------------------------------------
__global__ void k_ffn1(const unsigned short* x1b, const unsigned short* wf1_t,
                       const float* bf1, unsigned short* f1b) {
  int w    = blockIdx.x * (blockDim.x >> 5) + (threadIdx.x >> 5);
  int lane = threadIdx.x & 31;
  int h2 = lane >> 4, nm = lane & 15;
  int mt = w >> 6, nt = w & 63;
  int rowbase = mt * 16, colbase = nt * 16;
  v8f acc = {};
  for (int kt = 0; kt < 8; ++kt) {
    int k0 = kt * 32;
    v16bf a = load_frag_a(x1b + (rowbase + nm) * 256 + k0, h2);
    v16bf b = load_frag_b(wf1_t + (colbase + nm) * 256 + k0, h2);
    acc = wmma_bf16(a, b, acc);
  }
  int o = colbase + nm;
  float bias = bf1[o];
#pragma unroll
  for (int e = 0; e < 8; ++e) {
    int r = rowbase + e + 8 * h2;
    float val = acc[e] + bias;
    f1b[r * 1024 + o] = f2b(val > 0.f ? val : 0.f);
  }
}

// ---------------------------------------------------------------------------
// Kernel 9: FFN second linear + bias + residual (pre-LN2) fp32.
// Prefetch one K-tile ahead.
// ---------------------------------------------------------------------------
__global__ void k_ffn2(const unsigned short* f1b, const unsigned short* wf2_t,
                       const float* bf2, const float* x1, float* y2) {
  int w    = blockIdx.x * (blockDim.x >> 5) + (threadIdx.x >> 5);
  int lane = threadIdx.x & 31;
  int h2 = lane >> 4, nm = lane & 15;
  int mt = w >> 4, nt = w & 15;
  int rowbase = mt * 16, colbase = nt * 16;
  const unsigned short* arow = f1b + (rowbase + nm) * 1024;
  const unsigned short* brow = wf2_t + (colbase + nm) * 1024;
  v8f acc = {};
  for (int kt = 0; kt < 32; ++kt) {
    int k0 = kt * 32;
    if (kt + 1 < 32) {
      __builtin_prefetch(arow + k0 + 32, 0, 0);
      __builtin_prefetch(brow + k0 + 32, 0, 0);
    }
    v16bf a = load_frag_a(arow + k0, h2);
    v16bf b = load_frag_b(brow + k0, h2);
    acc = wmma_bf16(a, b, acc);
  }
  int o = colbase + nm;
  float bias = bf2[o];
#pragma unroll
  for (int e = 0; e < 8; ++e) {
    int r = rowbase + e + 8 * h2;
    y2[r * 256 + o] = acc[e] + bias + x1[r * 256 + o];
  }
}

// ---------------------------------------------------------------------------
extern "C" void kernel_launch(void* const* d_in, const int* in_sizes, int n_in,
                              void* d_out, int out_size, void* d_ws, size_t ws_size,
                              hipStream_t stream) {
  (void)in_sizes; (void)n_in; (void)out_size; (void)ws_size;

  InPtrs in;
  for (int i = 0; i < 24; ++i) in.p[i] = (const float*)d_in[i];

  char* base = (char*)d_ws;
  size_t off = 0;
  auto alloc = [&](size_t bytes) -> void* {
    void* p = base + off;
    off = (off + bytes + 255) & ~(size_t)255;
    return p;
  };
  unsigned short* wqkvr_t = (unsigned short*)alloc(1024 * 128 * 2);
  float*          bqkvr   = (float*)alloc(1024 * 4);
  unsigned short* wo_t    = (unsigned short*)alloc(256 * 256 * 2);
  unsigned short* wf1_t   = (unsigned short*)alloc(1024 * 256 * 2);
  unsigned short* wf2_t   = (unsigned short*)alloc(256 * 1024 * 2);
  unsigned short* wp2h_t  = (unsigned short*)alloc(16 * 256 * 2);
  float*          bp2h    = (float*)alloc(16 * 4);
  unsigned short* xb      = (unsigned short*)alloc(1024 * 128 * 2);
  unsigned short* qb      = (unsigned short*)alloc(1024 * 256 * 2);
  unsigned short* kb      = (unsigned short*)alloc(1024 * 256 * 2);
  unsigned short* vt      = (unsigned short*)alloc(256 * 1024 * 2);
  float*          ident   = (float*)alloc(1024 * 256 * 4);
  float*          S       = (float*)alloc((size_t)8 * 1024 * 1024 * 4);
  unsigned short* attnb   = (unsigned short*)alloc((size_t)8 * 1024 * 1024 * 2);
  unsigned short* ob      = (unsigned short*)alloc(1024 * 256 * 2);
  float*          x1raw   = (float*)alloc(1024 * 256 * 4);
  float*          x1      = (float*)alloc(1024 * 256 * 4);
  unsigned short* x1b     = (unsigned short*)alloc(1024 * 256 * 2);
  unsigned short* f1b     = (unsigned short*)alloc((size_t)1024 * 1024 * 2);
  float*          y2      = (float*)alloc(1024 * 256 * 4);

  dim3 blk(256);  // 8 waves of 32 on CDNA5
  k_prep   <<<2048, blk, 0, stream>>>(in, wqkvr_t, bqkvr, wo_t, wf1_t, wf2_t,
                                      wp2h_t, bp2h, xb);
  k_pe     <<<8192, blk, 0, stream>>>(in.p[1], in.p[8], in.p[9], wp2h_t, bp2h, S);
  k_qkvr   <<<512,  blk, 0, stream>>>(xb, wqkvr_t, bqkvr, qb, kb, vt, ident);
  k_qk     <<<4096, blk, 0, stream>>>(qb, kb, S);
  k_softmax<<<8192, blk, 0, stream>>>(S, attnb);
  k_av     <<<128,  blk, 0, stream>>>(attnb, vt, ob);
  k_oproj  <<<128,  blk, 0, stream>>>(ob, wo_t, in.p[13], ident, x1raw);
  k_ln     <<<1024, blk, 0, stream>>>(x1raw, in.p[16], in.p[17], x1, x1b);
  k_ffn1   <<<512,  blk, 0, stream>>>(x1b, wf1_t, in.p[21], f1b);
  k_ffn2   <<<128,  blk, 0, stream>>>(f1b, wf2_t, in.p[23], x1, y2);
  k_ln     <<<1024, blk, 0, stream>>>(y2, in.p[18], in.p[19], (float*)d_out, nullptr);
}
